// AD_61881888801137
// MI455X (gfx1250) — matrix operand order repaired and checked by reading, not compiled, not run on hardware
//
#include <hip/hip_runtime.h>
#include <hip/hip_bf16.h>
#include <stdint.h>

typedef __attribute__((ext_vector_type(16))) _Float16 v16h;
typedef __attribute__((ext_vector_type(8)))  _Float16 h8;
typedef __attribute__((ext_vector_type(8)))  float    v8f;
typedef int v4i __attribute__((vector_size(4 * sizeof(int))));

#define AS1 __attribute__((address_space(1)))
#define AS3 __attribute__((address_space(3)))

#define NB   16
#define NCTX 100
#define ND   1024
#define NH   16
#define NNL  4
#define NDFF 4096
#define NG   9
#define NNA  5
#define NLEN 301          // L
#define NLP  320          // padded L for PV K-dim
#define MROWS (NB * NLEN) // 4816

#if defined(__has_builtin)
#if __has_builtin(__builtin_amdgcn_global_load_async_to_lds_b128) && \
    __has_builtin(__builtin_amdgcn_s_wait_asynccnt)
#define USE_ASYNC_LDS 1
#endif
#endif

// ---------------------------------------------------------------------------
// WMMA helpers (gfx1250, wave32, 16x16x32 f16 -> f32)
// ---------------------------------------------------------------------------
__device__ __forceinline__ v8f wmma_f16(v16h a, v16h b, v8f c) {
  return __builtin_amdgcn_wmma_f32_16x16x32_f16(false, a, false, b, (short)0, c,
                                                false, false);
}

// Branch-free zero mask: multiply by 0/1 splat (cannot be folded without
// fast-math, so the feeding loads stay unconditional -> no exec branches).
__device__ __forceinline__ v16h mask16(v16h r, bool ok) {
  _Float16 m = ok ? (_Float16)1.0f : (_Float16)0.0f;
#pragma unroll
  for (int i = 0; i < 16; i++) r[i] *= m;
  return r;
}

// A fragment, row-major A[M,K]: lane m=lane&15 (row), g=lane>>4.
// half j -> K = k0+16g+j.
__device__ __forceinline__ v16h ldA_u(const _Float16* A, long lda, int row0,
                                      int k0) {
  int lane = threadIdx.x & 31;
  int m = lane & 15;
  int g = lane >> 4;
  const h8* p = (const h8*)(A + (long)(row0 + m) * lda + k0 + 16 * g);
  h8 lo = p[0];
  h8 hi = p[1];  // +8 halves
  return __builtin_shufflevector(lo, hi, 0, 1, 2, 3, 4, 5, 6, 7, 8, 9, 10, 11,
                                 12, 13, 14, 15);
}

// Guarded (branch-free): clamp row (always-safe address), mask with mul.
__device__ __forceinline__ v16h ldA_g(const _Float16* A, long lda, int row0,
                                      int k0, int rowLim) {
  int lane = threadIdx.x & 31;
  int m = lane & 15;
  int g = lane >> 4;
  int row = row0 + m;
  bool ok = row < rowLim;
  int rc = ok ? row : (rowLim - 1);
  const h8* p = (const h8*)(A + (long)rc * lda + k0 + 16 * g);
  h8 lo = p[0];
  h8 hi = p[1];
  v16h r = __builtin_shufflevector(lo, hi, 0, 1, 2, 3, 4, 5, 6, 7, 8, 9, 10,
                                   11, 12, 13, 14, 15);
  return mask16(r, ok);
}

// A fragment from an LDS-staged 16x64 tile (row stride 64 halves).
__device__ __forceinline__ v16h ldA_lds(const _Float16* S, int k0) {
  int lane = threadIdx.x & 31;
  int m = lane & 15;
  int g = lane >> 4;
  const h8* p = (const h8*)(S + m * 64 + k0 + 16 * g);
  h8 lo = p[0];
  h8 hi = p[1];
  return __builtin_shufflevector(lo, hi, 0, 1, 2, 3, 4, 5, 6, 7, 8, 9, 10, 11,
                                 12, 13, 14, 15);
}

// B fragment from K-contiguous (column-major) storage: element (k,n) at
// Kt[n*ld + k]. lane n=lane&15 (col), g=lane>>4; half j -> K = k0+16g+j.
__device__ __forceinline__ v16h ldBt_u(const _Float16* Kt, long ld, int k0,
                                       int col0) {
  int lane = threadIdx.x & 31;
  int n = lane & 15;
  int kb = (lane >> 4) * 16;
  const h8* p = (const h8*)(Kt + (long)(col0 + n) * ld + k0 + kb);
  h8 lo = p[0];
  h8 hi = p[1];
  return __builtin_shufflevector(lo, hi, 0, 1, 2, 3, 4, 5, 6, 7, 8, 9, 10, 11,
                                 12, 13, 14, 15);
}

__device__ __forceinline__ v16h ldBt_g(const _Float16* Kt, long ld, int k0,
                                       int col0, int colLim) {
  int lane = threadIdx.x & 31;
  int n = lane & 15;
  int kb = (lane >> 4) * 16;
  int col = col0 + n;
  bool ok = col < colLim;
  int cc = ok ? col : (colLim - 1);
  const h8* p = (const h8*)(Kt + (long)cc * ld + k0 + kb);
  h8 lo = p[0];
  h8 hi = p[1];
  v16h r = __builtin_shufflevector(lo, hi, 0, 1, 2, 3, 4, 5, 6, 7, 8, 9, 10,
                                   11, 12, 13, 14, 15);
  return mask16(r, ok);
}

// B fragment from WMMA-packed weights: layout [N/16][K/16][16 n][16 k].
__device__ __forceinline__ v16h ldBp(const _Float16* Bp, int Kdim, int k0,
                                     int colTile) {
  int lane = threadIdx.x & 31;
  int n = lane & 15;
  int g = lane >> 4;
  long off = ((long)(colTile >> 4) * (Kdim >> 4) + (k0 >> 4) + g) * 256 +
             (long)n * 16;
  const h8* p = (const h8*)(Bp + off);
  h8 lo = p[0];
  h8 hi = p[1];
  return __builtin_shufflevector(lo, hi, 0, 1, 2, 3, 4, 5, 6, 7, 8, 9, 10, 11,
                                 12, 13, 14, 15);
}

__device__ __forceinline__ float gelu_new(float x) {
  float x3 = x * x * x;
  return 0.5f * x * (1.0f + tanhf(0.7978845608028654f * (x + 0.044715f * x3)));
}

// Stage one 16x64 A tile (2KB) into LDS: 128 threads x one b128 each.
// Async (ASYNCcnt path) when available, else synchronous copy.
__device__ __forceinline__ void stage_A(const _Float16* A, long K, int row0,
                                        int kc, _Float16* sbuf) {
  int tid = threadIdx.x;          // 0..127
  int row = tid >> 3;             // 16 rows
  int seg = tid & 7;              // 8 x 8 halves
  const _Float16* g = A + (long)(row0 + row) * K + kc + seg * 8;
  _Float16* l = sbuf + row * 64 + seg * 8;
#if defined(USE_ASYNC_LDS)
  __builtin_amdgcn_global_load_async_to_lds_b128((AS1 v4i*)g, (AS3 v4i*)l, 0,
                                                 0);
#else
  *(h8*)l = *(const h8*)g;
#endif
}

__device__ __forceinline__ void wait_stage() {
#if defined(USE_ASYNC_LDS)
  __builtin_amdgcn_s_wait_asynccnt(0);
#endif
  __syncthreads();
}

// ---------------------------------------------------------------------------
// Generic WMMA GEMM: C[M,N] = A[M,K](f16) @ Bp(packed f16) + bias
// MODE 0: store f16   MODE 1: resid(f32) += val   MODE 2: gelu -> f16
// block = 128 threads (4 waves); wave computes 16 rows x 64 cols.
// A tile double-buffered through LDS via async global->LDS DMA.
// Requires M%16==0, K%64==0, N%256==0.
// ---------------------------------------------------------------------------
template <int MODE>
__global__ void gemm_wmma(const _Float16* __restrict__ A,
                          const _Float16* __restrict__ Bp,
                          const float* __restrict__ bias,
                          _Float16* __restrict__ out16,
                          float* __restrict__ resid, int M, int N, int K) {
  __shared__ _Float16 sA[2][16 * 64];
  int lane = threadIdx.x & 31;
  int wave = threadIdx.x >> 5;
  int row0 = blockIdx.x * 16;
  int col0 = blockIdx.y * 256 + wave * 64;

  v8f zero;
#pragma unroll
  for (int r = 0; r < 8; r++) zero[r] = 0.0f;
  v8f acc[4] = {zero, zero, zero, zero};

  int nc = K >> 6;  // 64-wide K chunks
  stage_A(A, K, row0, 0, sA[0]);
  for (int c = 0; c < nc; c++) {
    wait_stage();  // chunk c landed (all waves), buffers free
    if (c + 1 < nc) stage_A(A, K, row0, (c + 1) << 6, sA[(c + 1) & 1]);
    const _Float16* Sc = sA[c & 1];
    int kbase = c << 6;
#pragma unroll
    for (int kk = 0; kk < 64; kk += 32) {
      v16h a = ldA_lds(Sc, kk);
#pragma unroll
      for (int t = 0; t < 4; t++) {
        v16h b = ldBp(Bp, K, kbase + kk, col0 + t * 16);
        acc[t] = wmma_f16(a, b, acc[t]);
      }
    }
  }

#pragma unroll
  for (int t = 0; t < 4; t++) {
    int col = col0 + t * 16 + (lane & 15);
    float bb = bias[col];
#pragma unroll
    for (int r = 0; r < 8; r++) {
      int row = row0 + r + ((lane >> 4) << 3);
      float v = acc[t][r] + bb;
      if (MODE == 0) {
        out16[(long)row * N + col] = (_Float16)v;
      } else if (MODE == 1) {
        resid[(long)row * N + col] += v;
      } else {
        out16[(long)row * N + col] = (_Float16)gelu_new(v);
      }
    }
  }
}

// ---------------------------------------------------------------------------
// Embedding assembly: h[b,l,d] = token + wpe[l]
// ---------------------------------------------------------------------------
__global__ void embed_kernel(const int* __restrict__ states,
                             const int* __restrict__ actions,
                             const float* __restrict__ rewards,
                             const int* __restrict__ qstates,
                             const float* __restrict__ wpe,
                             const float* __restrict__ est,
                             const float* __restrict__ eact,
                             const float* __restrict__ erw,
                             const float* __restrict__ erb,
                             float* __restrict__ h) {
  long idx = (long)blockIdx.x * blockDim.x + threadIdx.x;
  long total = (long)NB * NLEN * ND;
  if (idx >= total) return;
  int d = (int)(idx % ND);
  long rem = idx / ND;
  int l = (int)(rem % NLEN);
  int b = (int)(rem / NLEN);
  float val;
  if (l == NLEN - 1) {
    int qid = qstates[b * 2] * NG + qstates[b * 2 + 1];
    val = est[(long)qid * ND + d];
  } else {
    int t = l / 3, r = l % 3;
    if (r == 0) {
      int sid = states[(b * NCTX + t) * 2] * NG + states[(b * NCTX + t) * 2 + 1];
      val = est[(long)sid * ND + d];
    } else if (r == 1) {
      val = eact[(long)actions[b * NCTX + t] * ND + d];
    } else {
      val = rewards[b * NCTX + t] * erw[d] + erb[d];
    }
  }
  h[idx] = val + wpe[(long)l * ND + d];
}

// ---------------------------------------------------------------------------
// LayerNorm over D=1024, one row per block (256 threads), f16 output
// ---------------------------------------------------------------------------
__global__ void ln_kernel(const float* __restrict__ X,
                          const float* __restrict__ g,
                          const float* __restrict__ bt,
                          _Float16* __restrict__ out) {
  __shared__ float red[256];
  int row = blockIdx.x;
  int tid = threadIdx.x;
  const float* xr = X + (long)row * ND;
  float vals[4], s = 0.f, ss = 0.f;
#pragma unroll
  for (int i = 0; i < 4; i++) {
    float v = xr[tid + i * 256];
    vals[i] = v; s += v; ss += v * v;
  }
  red[tid] = s; __syncthreads();
  for (int st = 128; st > 0; st >>= 1) {
    if (tid < st) red[tid] += red[tid + st];
    __syncthreads();
  }
  float mean = red[0] / (float)ND;
  __syncthreads();
  red[tid] = ss; __syncthreads();
  for (int st = 128; st > 0; st >>= 1) {
    if (tid < st) red[tid] += red[tid + st];
    __syncthreads();
  }
  float var = red[0] / (float)ND - mean * mean;
  float rstd = rsqrtf(var + 1e-5f);
#pragma unroll
  for (int i = 0; i < 4; i++) {
    int j = tid + i * 256;
    out[(long)row * ND + j] = (_Float16)((vals[i] - mean) * rstd * g[j] + bt[j]);
  }
}

// ---------------------------------------------------------------------------
// Attention scores: one wave per 16x16 tile; skip fully-masked tiles.
// qkv layout: [B*L, 3*D] f16 (q | k | v), head hh at col offset hh*64.
// ---------------------------------------------------------------------------
__global__ void attn_scores(const _Float16* __restrict__ qkv,
                            float* __restrict__ scores) {
  int bh = blockIdx.z;
  int b = bh >> 4, hh = bh & 15;
  int row0 = blockIdx.x * 16, col0 = blockIdx.y * 16;
  if (col0 > row0 + 15) return;  // causal: whole tile masked, never read
  const _Float16* qb = qkv + ((long)b * NLEN) * (3 * ND) + hh * 64;
  const _Float16* kb = qkv + ((long)b * NLEN) * (3 * ND) + ND + hh * 64;
  int lane = threadIdx.x & 31;
  v8f acc;
#pragma unroll
  for (int r = 0; r < 8; r++) acc[r] = 0.0f;
  bool interior = (row0 + 16 <= NLEN) && (col0 + 16 <= NLEN);
  if (interior) {
#pragma unroll
    for (int k = 0; k < 64; k += 32) {
      v16h a = ldA_u(qb, 3 * ND, row0, k);
      v16h bf = ldBt_u(kb, 3 * ND, k, col0);
      acc = wmma_f16(a, bf, acc);
    }
    float* sp = scores + ((long)bh * NLEN + row0) * NLEN + col0 + (lane & 15);
#pragma unroll
    for (int r = 0; r < 8; r++) {
      int m = r + ((lane >> 4) << 3);
      sp[(long)m * NLEN] = acc[r] * 0.125f;
    }
  } else {
#pragma unroll
    for (int k = 0; k < 64; k += 32) {
      v16h a = ldA_g(qb, 3 * ND, row0, k, NLEN);
      v16h bf = ldBt_g(kb, 3 * ND, k, col0, NLEN);
      acc = wmma_f16(a, bf, acc);
    }
    int n = col0 + (lane & 15);
#pragma unroll
    for (int r = 0; r < 8; r++) {
      int m = row0 + r + ((lane >> 4) << 3);
      if (m < NLEN && n < NLEN)
        scores[((long)bh * NLEN + m) * NLEN + n] = acc[r] * 0.125f;
    }
  }
}

// Row softmax with causal mask; writes f16 probs padded to NLP cols (zeros).
__global__ void softmax_kernel(const float* __restrict__ scores,
                               _Float16* __restrict__ probs) {
  __shared__ float red[128];
  long row = blockIdx.x;
  int q = (int)(row % NLEN);
  long bh = row / NLEN;
  int tid = threadIdx.x;
  const float* sr = scores + ((long)bh * NLEN + q) * NLEN;
  int n = q + 1;
  float mx = -3.4e38f;
  for (int j = tid; j < n; j += 128) mx = fmaxf(mx, sr[j]);
  red[tid] = mx; __syncthreads();
  for (int st = 64; st > 0; st >>= 1) {
    if (tid < st) red[tid] = fmaxf(red[tid], red[tid + st]);
    __syncthreads();
  }
  mx = red[0]; __syncthreads();
  float s = 0.f;
  for (int j = tid; j < n; j += 128) s += __expf(sr[j] - mx);
  red[tid] = s; __syncthreads();
  for (int st = 64; st > 0; st >>= 1) {
    if (tid < st) red[tid] += red[tid + st];
    __syncthreads();
  }
  float inv = 1.0f / red[0];
  _Float16* pr = probs + ((long)bh * NLEN + q) * NLP;
  for (int j = tid; j < NLP; j += 128)
    pr[j] = (j < n) ? (_Float16)(__expf(sr[j] - mx) * inv) : (_Float16)0.0f;
}

// Repack V transposed + zero-padded: vT[bh][d][k] (k padded to NLP)
__global__ void repack_v(const _Float16* __restrict__ qkv,
                         _Float16* __restrict__ vT) {
  long idx = (long)blockIdx.x * blockDim.x + threadIdx.x;
  long total = (long)NB * NH * 64 * NLP;
  if (idx >= total) return;
  int k = (int)(idx % NLP);
  long rem = idx / NLP;
  int d = (int)(rem % 64);
  int bh = (int)(rem / 64);
  int b = bh >> 4, hh = bh & 15;
  _Float16 v = (_Float16)0.0f;
  if (k < NLEN) v = qkv[((long)b * NLEN + k) * (3 * ND) + 2 * ND + hh * 64 + d];
  vT[idx] = v;
}

// PV: attn[q,d] = probs[q,:] @ v[:,d]; one wave per 16-row stripe (64 cols).
__global__ void attn_pv(const _Float16* __restrict__ probs,
                        const _Float16* __restrict__ vT,
                        _Float16* __restrict__ attnout) {
  int bh = blockIdx.y;
  int b = bh >> 4, hh = bh & 15;
  int row0 = blockIdx.x * 16;
  const _Float16* Ap = probs + (long)bh * NLEN * NLP;
  const _Float16* Bv = vT + (long)bh * 64 * NLP;  // [d][k], k contiguous
  int lane = threadIdx.x & 31;
  v8f zero;
#pragma unroll
  for (int r = 0; r < 8; r++) zero[r] = 0.0f;
  v8f acc[4] = {zero, zero, zero, zero};
  bool interior = (row0 + 16 <= NLEN);
  if (interior) {
#pragma unroll 2
    for (int k = 0; k < NLP; k += 32) {
      v16h a = ldA_u(Ap, NLP, row0, k);
#pragma unroll
      for (int t = 0; t < 4; t++) {
        v16h bf = ldBt_u(Bv, NLP, k, t * 16);
        acc[t] = wmma_f16(a, bf, acc[t]);
      }
    }
    _Float16* op = attnout + ((long)b * NLEN + row0) * ND + hh * 64;
#pragma unroll
    for (int t = 0; t < 4; t++) {
      int col = t * 16 + (lane & 15);
#pragma unroll
      for (int r = 0; r < 8; r++) {
        int m = r + ((lane >> 4) << 3);
        op[(long)m * ND + col] = (_Float16)acc[t][r];
      }
    }
  } else {
#pragma unroll 2
    for (int k = 0; k < NLP; k += 32) {
      v16h a = ldA_g(Ap, NLP, row0, k, NLEN);
#pragma unroll
      for (int t = 0; t < 4; t++) {
        v16h bf = ldBt_u(Bv, NLP, k, t * 16);
        acc[t] = wmma_f16(a, bf, acc[t]);
      }
    }
#pragma unroll
    for (int t = 0; t < 4; t++) {
      int col = hh * 64 + t * 16 + (lane & 15);
#pragma unroll
      for (int r = 0; r < 8; r++) {
        int m = row0 + r + ((lane >> 4) << 3);
        if (m < NLEN)
          attnout[((long)b * NLEN + m) * ND + col] = (_Float16)acc[t][r];
      }
    }
  }
}

// ---------------------------------------------------------------------------
// f32 weights -> f16 WMMA-packed: src[K,N] -> dst[N/16][K/16][16 n][16 k]
// ---------------------------------------------------------------------------
__global__ void cvt_pack(const float* __restrict__ src,
                         _Float16* __restrict__ dst, int K, int N) {
  long idx = (long)blockIdx.x * blockDim.x + threadIdx.x;
  long total = (long)K * N;
  if (idx >= total) return;
  int k = (int)(idx / N);
  int n = (int)(idx % N);
  long off = (((long)(n >> 4) * (K >> 4) + (k >> 4)) * 16 + (n & 15)) * 16 +
             (k & 15);
  dst[off] = (_Float16)src[idx];
}

// Logits at state-token positions: one thread per (row, class)
__global__ void logits_kernel(const _Float16* __restrict__ xf,
                              const float* __restrict__ pw,
                              const float* __restrict__ pb,
                              float* __restrict__ logits) {
  int idx = blockIdx.x * blockDim.x + threadIdx.x;
  int nrows = NB * (NCTX + 1);
  if (idx >= nrows * NNA) return;
  int row = idx / NNA, c = idx % NNA;
  int b = row / (NCTX + 1), t = row % (NCTX + 1);
  const _Float16* xr = xf + ((long)b * NLEN + 3 * t) * ND;
  float acc = pb[c];
  for (int d = 0; d < ND; d++) acc += (float)xr[d] * pw[d * NNA + c];
  logits[row * NNA + c] = acc;
}

// Per-row CE with label smoothing + correctness flag
__global__ void loss_kernel(const float* __restrict__ logits,
                            const int* __restrict__ actions,
                            const int* __restrict__ tact,
                            float* __restrict__ lossb,
                            float* __restrict__ accb) {
  int row = blockIdx.x * blockDim.x + threadIdx.x;
  int nrows = NB * (NCTX + 1);
  if (row >= nrows) return;
  int b = row / (NCTX + 1), t = row % (NCTX + 1);
  int tgt = (t < NCTX) ? actions[b * NCTX + t] : tact[b];
  float lg[NNA];
  float mx = -3.4e38f;
  int am = 0;
#pragma unroll
  for (int c = 0; c < NNA; c++) {
    lg[c] = logits[row * NNA + c];
    if (lg[c] > mx) { mx = lg[c]; am = c; }
  }
  float se = 0.f, sl = 0.f;
#pragma unroll
  for (int c = 0; c < NNA; c++) { se += __expf(lg[c] - mx); sl += lg[c]; }
  float lse = mx + logf(se);
  float nll = -(lg[tgt] - lse);
  float smooth = -(sl - (float)NNA * lse) / (float)NNA;
  lossb[row] = 0.9f * nll + 0.1f * smooth;
  accb[row] = (am == tgt) ? 1.0f : 0.0f;
}

__global__ void reduce_kernel(const float* __restrict__ lossb,
                              const float* __restrict__ accb,
                              float* __restrict__ out) {
  __shared__ float r1[256], r2[256];
  int tid = threadIdx.x;
  int n = NB * (NCTX + 1);
  float s1 = 0.f, s2 = 0.f;
  for (int i = tid; i < n; i += 256) { s1 += lossb[i]; s2 += accb[i]; }
  r1[tid] = s1; r2[tid] = s2; __syncthreads();
  for (int st = 128; st > 0; st >>= 1) {
    if (tid < st) { r1[tid] += r1[tid + st]; r2[tid] += r2[tid + st]; }
    __syncthreads();
  }
  if (tid == 0) {
    out[0] = r1[0] / (float)n;
    out[1] = r2[0] / (float)n;
  }
}

// ---------------------------------------------------------------------------
// Host launcher
// ---------------------------------------------------------------------------
static inline size_t alignup(size_t x) { return (x + 255) & ~(size_t)255; }

extern "C" void kernel_launch(void* const* d_in, const int* in_sizes, int n_in,
                              void* d_out, int out_size, void* d_ws,
                              size_t ws_size, hipStream_t stream) {
  (void)in_sizes; (void)n_in; (void)out_size; (void)ws_size;
  const int*   states   = (const int*)d_in[0];
  const int*   actions  = (const int*)d_in[1];
  const float* rewards  = (const float*)d_in[2];
  const int*   qstates  = (const int*)d_in[3];
  const int*   tact     = (const int*)d_in[4];
  const float* wpe      = (const float*)d_in[5];
  const float* est      = (const float*)d_in[6];
  const float* eact     = (const float*)d_in[7];
  const float* erw      = (const float*)d_in[8];
  const float* erb      = (const float*)d_in[9];
  const float* ln1_g    = (const float*)d_in[10];
  const float* ln1_b    = (const float*)d_in[11];
  const float* attn_w   = (const float*)d_in[12];
  const float* attn_b   = (const float*)d_in[13];
  const float* aproj_w  = (const float*)d_in[14];
  const float* aproj_b  = (const float*)d_in[15];
  const float* ln2_g    = (const float*)d_in[16];
  const float* ln2_b    = (const float*)d_in[17];
  const float* fc_w     = (const float*)d_in[18];
  const float* fc_b     = (const float*)d_in[19];
  const float* mproj_w  = (const float*)d_in[20];
  const float* mproj_b  = (const float*)d_in[21];
  const float* lnf_g    = (const float*)d_in[22];
  const float* lnf_b    = (const float*)d_in[23];
  const float* pred_w   = (const float*)d_in[24];
  const float* pred_b   = (const float*)d_in[25];

  uint8_t* p = (uint8_t*)d_ws;
  float*     hbuf    = (float*)p;     p += alignup((size_t)MROWS * ND * 4);
  _Float16*  x16     = (_Float16*)p;  p += alignup((size_t)MROWS * ND * 2);
  _Float16*  qkv16   = (_Float16*)p;  p += alignup((size_t)MROWS * 3 * ND * 2);
  _Float16*  aout16  = (_Float16*)p;  p += alignup((size_t)MROWS * ND * 2);
  _Float16*  mlp16   = (_Float16*)p;  p += alignup((size_t)MROWS * NDFF * 2);
  float*     scores  = (float*)p;     p += alignup((size_t)NB * NH * NLEN * NLEN * 4);
  _Float16*  probs16 = (_Float16*)p;  p += alignup((size_t)NB * NH * NLEN * NLP * 2);
  _Float16*  vT16    = (_Float16*)p;  p += alignup((size_t)NB * NH * 64 * NLP * 2);
  _Float16*  wqkv16  = (_Float16*)p;  p += alignup((size_t)ND * 3 * ND * 2);
  _Float16*  wproj16 = (_Float16*)p;  p += alignup((size_t)ND * ND * 2);
  _Float16*  wfc16   = (_Float16*)p;  p += alignup((size_t)ND * NDFF * 2);
  _Float16*  wmp16   = (_Float16*)p;  p += alignup((size_t)NDFF * ND * 2);
  float*     logits  = (float*)p;     p += alignup((size_t)NB * (NCTX + 1) * NNA * 4);
  float*     lossb   = (float*)p;     p += alignup((size_t)NB * (NCTX + 1) * 4);
  float*     accb    = (float*)p;     p += alignup((size_t)NB * (NCTX + 1) * 4);

  // 1. Embedding assembly -> h[B*L, D]
  {
    long n = (long)MROWS * ND;
    embed_kernel<<<(unsigned)((n + 255) / 256), 256, 0, stream>>>(
        states, actions, rewards, qstates, wpe, est, eact, erw, erb, hbuf);
  }

  const int LT = (NLEN + 15) / 16;  // 19 row/col tiles
  for (int i = 0; i < NNL; i++) {
    // --- LN1 -> x16
    ln_kernel<<<MROWS, 256, 0, stream>>>(hbuf, ln1_g + (long)i * ND,
                                         ln1_b + (long)i * ND, x16);
    // --- QKV GEMM (WMMA)
    {
      long n = (long)ND * 3 * ND;
      cvt_pack<<<(unsigned)((n + 255) / 256), 256, 0, stream>>>(
          attn_w + (long)i * n, wqkv16, ND, 3 * ND);
      gemm_wmma<0><<<dim3(MROWS / 16, (3 * ND) / 256), 128, 0, stream>>>(
          x16, wqkv16, attn_b + (long)i * 3 * ND, qkv16, nullptr,
          MROWS, 3 * ND, ND);
    }
    // --- attention: scores (WMMA) / softmax / V repack / PV (WMMA)
    attn_scores<<<dim3(LT, LT, NB * NH), 32, 0, stream>>>(qkv16, scores);
    softmax_kernel<<<NB * NH * NLEN, 128, 0, stream>>>(scores, probs16);
    {
      long n = (long)NB * NH * 64 * NLP;
      repack_v<<<(unsigned)((n + 255) / 256), 256, 0, stream>>>(qkv16, vT16);
    }
    attn_pv<<<dim3(LT, NB * NH), 32, 0, stream>>>(probs16, vT16, aout16);
    // --- attention output projection + residual
    {
      long n = (long)ND * ND;
      cvt_pack<<<(unsigned)((n + 255) / 256), 256, 0, stream>>>(
          aproj_w + (long)i * n, wproj16, ND, ND);
      gemm_wmma<1><<<dim3(MROWS / 16, ND / 256), 128, 0, stream>>>(
          aout16, wproj16, aproj_b + (long)i * ND, nullptr, hbuf,
          MROWS, ND, ND);
    }
    // --- LN2 -> x16
    ln_kernel<<<MROWS, 256, 0, stream>>>(hbuf, ln2_g + (long)i * ND,
                                         ln2_b + (long)i * ND, x16);
    // --- FC + gelu_new
    {
      long n = (long)ND * NDFF;
      cvt_pack<<<(unsigned)((n + 255) / 256), 256, 0, stream>>>(
          fc_w + (long)i * n, wfc16, ND, NDFF);
      gemm_wmma<2><<<dim3(MROWS / 16, NDFF / 256), 128, 0, stream>>>(
          x16, wfc16, fc_b + (long)i * NDFF, mlp16, nullptr,
          MROWS, NDFF, ND);
    }
    // --- MLP projection + residual
    {
      long n = (long)NDFF * ND;
      cvt_pack<<<(unsigned)((n + 255) / 256), 256, 0, stream>>>(
          mproj_w + (long)i * n, wmp16, NDFF, ND);
      gemm_wmma<1><<<dim3(MROWS / 16, ND / 256), 128, 0, stream>>>(
          mlp16, wmp16, mproj_b + (long)i * ND, nullptr, hbuf,
          MROWS, ND, NDFF);
    }
  }

  // Final LN -> x16
  ln_kernel<<<MROWS, 256, 0, stream>>>(hbuf, lnf_g, lnf_b, x16);
  // Logits at state-token positions
  {
    int n = NB * (NCTX + 1) * NNA;
    logits_kernel<<<(n + 255) / 256, 256, 0, stream>>>(x16, pred_w, pred_b,
                                                       logits);
  }
  // Loss + accuracy
  {
    int n = NB * (NCTX + 1);
    loss_kernel<<<(n + 255) / 256, 256, 0, stream>>>(logits, actions, tact,
                                                     lossb, accb);
  }
  reduce_kernel<<<1, 256, 0, stream>>>(lossb, accb, (float*)d_out);
}
